// MSCrossAttnBlockDAUL_5549097746875
// MI455X (gfx1250) — matrix operand order repaired and checked by reading, not compiled, not run on hardware
//
#include <hip/hip_runtime.h>
#include <math.h>

// ---------------------------------------------------------------------------
// MSCrossAttnBlock for MI455X (gfx1250, wave32, WMMA bf16 16x16x32, f32 acc)
// Weights pre-packed to transposed bf16 so every B fragment is 2x b128 loads.
// ---------------------------------------------------------------------------

typedef __attribute__((ext_vector_type(16))) __bf16 v16bf;
typedef __attribute__((ext_vector_type(8)))  float  v8f;

#define D_MODEL 256
#define N_HEADS 16
#define DH      16
#define NPTS    4
#define LQ      1024
#define NB      8
#define S_TOT   21504   // 128^2 + 64^2 + 32^2

struct Stage2 {
  const __bf16* Wt;    // prepacked: nOut x 256 bf16 (i.e. W^T)
  const float*  bias;  // nOut (may be null)
  void*         out;   // rows x nOut
  int           nOut;  // multiple of 64
  int           outBF16;
};

union Frag { v16bf v; uint4 q[2]; unsigned u[8]; };

// Build all 8 K-fragments (K=256) of the 16x32 bf16 A layout from LDS.
// Layout (CDNA5 ISA): lane<16: V0..3 K=0..7, V4..7 K=16..23; lane>=16: +8.
// Each half is 4 consecutive dwords -> 2x ds_read_b128 per fragment.
__device__ inline void build_afrags(Frag* af8, const unsigned* aB32,
                                    int l16, int halfsel)
{
  #pragma unroll
  for (int kt = 0; kt < 8; ++kt) {
    const uint4* ap = (const uint4*)(aB32 + (l16 * 128 + kt * 16 + halfsel * 4));
    af8[kt].q[0] = ap[0];   // K = kb .. kb+7
    af8[kt].q[1] = ap[2];   // K = kb+16 .. kb+23
  }
}

// B fragment (32x16 bf16): lanes0-15 hold K=0..7,16..23 of column l16, lanes16-31
// K=8..15,24..31 (same order as A). With W^T prepack: 2x b128 loads.
__device__ inline Frag load_bfrag(const __bf16* Wt, int col, int kt, int halfsel)
{
  Frag b;
  const uint4* wp = (const uint4*)((const unsigned*)Wt +
                                   ((size_t)col * 128 + kt * 16 + halfsel * 4));
  b.q[0] = wp[0];
  b.q[1] = wp[2];
  return b;
}

// ---------------------------------------------------------------------------
// Fused kernel:  x = A(16x256 stripe) @ W1 + b1
//                optionally write x to Xout (row-remapped)
//                optionally LayerNorm(x)*g+beta
//                then up to 3 projections  out_s = ln(x) @ W_s + b_s
// One workgroup = 128 threads = 4 waves; each wave owns 4 16x16 N-tiles.
// orow = (gr / rpbIn) * rpbOut + outRowBase + gr % rpbIn
// ---------------------------------------------------------------------------
__global__ __launch_bounds__(128)
void fused_ln_gemm(const float* __restrict__ A,
                   const __bf16* __restrict__ W1t, const float* __restrict__ b1,
                   const float* __restrict__ g,  const float* __restrict__ beta,
                   float* __restrict__ Xout, int doLN,
                   Stage2 s0, Stage2 s1, Stage2 s2, int nStage2,
                   int rpbIn, int rpbOut, int outRowBase)
{
  __shared__ __align__(16) __bf16 aBuf[16 * 256];  // A (bf16), later LN(x)
  __shared__ __align__(16) float  xBuf[16 * 256];  // stage-1 result (f32)
  __shared__ float muS[16], rsS[16];

  const int t       = threadIdx.x;
  const int wave    = t >> 5;
  const int lane    = t & 31;
  const int l16     = lane & 15;
  const int halfsel = lane >> 4;
  const int row0    = blockIdx.x * 16;

  // ---- load A stripe (16x256 f32) -> bf16 in LDS; prefetch next stripe ----
  {
    const float4* Ap = (const float4*)(A + (size_t)row0 * 256);
    for (int i = t; i < 16 * 256 / 4; i += 128) {
      float4 f4 = Ap[i];
      __bf16* d = aBuf + i * 4;
      d[0] = (__bf16)f4.x; d[1] = (__bf16)f4.y;
      d[2] = (__bf16)f4.z; d[3] = (__bf16)f4.w;
    }
    if (blockIdx.x + 1 < gridDim.x) {
      const float* An = A + (size_t)(row0 + 16) * 256;
      for (int i = t; i < 64; i += 128)
        __builtin_prefetch(An + i * 64, 0, 1);   // global_prefetch_b8
    }
  }
  __syncthreads();

  const unsigned* aB32 = (const unsigned*)aBuf;

  // ---- stage 1: X = A @ W1 ----
  Frag af8[8];
  build_afrags(af8, aB32, l16, halfsel);

  v8f acc[4] = {};
  #pragma unroll
  for (int kt = 0; kt < 8; ++kt) {
    #pragma unroll
    for (int i = 0; i < 4; ++i) {
      const int col = (wave * 4 + i) * 16 + l16;
      Frag bf = load_bfrag(W1t, col, kt, halfsel);
      acc[i] = __builtin_amdgcn_wmma_f32_16x16x32_bf16(
                 false, af8[kt].v, false, bf.v, (short)0, acc[i], false, false);
    }
  }

  // ---- scatter C (+bias) into xBuf ----
  #pragma unroll
  for (int i = 0; i < 4; ++i) {
    const int col = (wave * 4 + i) * 16 + l16;
    const float bb = b1 ? b1[col] : 0.f;
    #pragma unroll
    for (int v = 0; v < 8; ++v) {
      const int m = halfsel * 8 + v;   // C/D layout: VGPR v -> M = v (+8)
      xBuf[m * 256 + col] = acc[i][v] + bb;
    }
  }
  __syncthreads();

  // ---- optional write-out of x ----
  if (Xout) {
    for (int i = t; i < 16 * 256 / 4; i += 128) {
      const int r  = (i * 4) >> 8;
      const int c  = (i * 4) & 255;
      const int gr = row0 + r;
      const int orow = (gr / rpbIn) * rpbOut + outRowBase + (gr % rpbIn);
      ((float4*)Xout)[((size_t)orow * 256 + c) >> 2] = ((const float4*)xBuf)[i];
    }
  }

  if (nStage2 == 0) return;

  // ---- LayerNorm + affine -> bf16 in aBuf ----
  if (doLN) {
    if (t < 16) {
      const float* xr = xBuf + t * 256;
      float s = 0.f, ss = 0.f;
      for (int c = 0; c < 256; ++c) { float x = xr[c]; s += x; ss += x * x; }
      const float m  = s * (1.f / 256.f);
      const float va = ss * (1.f / 256.f) - m * m;
      muS[t] = m;
      rsS[t] = rsqrtf(va + 1e-6f);
    }
    __syncthreads();
    for (int i = t; i < 16 * 256; i += 128) {
      const int r = i >> 8, c = i & 255;
      aBuf[i] = (__bf16)(((xBuf[i] - muS[r]) * rsS[r]) * g[c] + beta[c]);
    }
    __syncthreads();
  }

  // ---- stage 2: projections of LN(x); A fragments hoisted in registers ----
  build_afrags(af8, aB32, l16, halfsel);

  Stage2 ss2[3] = { s0, s1, s2 };
  for (int sI = 0; sI < nStage2; ++sI) {
    const Stage2 st = ss2[sI];
    const int nT = st.nOut >> 4;
    for (int nt = wave; nt < nT; nt += 4) {
      const int col = nt * 16 + l16;
      v8f c2 = {};
      #pragma unroll
      for (int kt = 0; kt < 8; ++kt) {
        Frag bf = load_bfrag(st.Wt, col, kt, halfsel);
        c2 = __builtin_amdgcn_wmma_f32_16x16x32_bf16(
               false, af8[kt].v, false, bf.v, (short)0, c2, false, false);
      }
      const float bb = st.bias ? st.bias[col] : 0.f;
      #pragma unroll
      for (int v = 0; v < 8; ++v) {
        const int m  = halfsel * 8 + v;
        const int gr = row0 + m;
        const int orow = (gr / rpbIn) * rpbOut + outRowBase + (gr % rpbIn);
        const float val = c2[v] + bb;
        if (st.outBF16)
          ((__bf16*)st.out)[(size_t)orow * st.nOut + col] = (__bf16)val;
        else
          ((float*)st.out)[(size_t)orow * st.nOut + col] = val;
      }
    }
  }
}

// ---------------------------------------------------------------------------
// Weight prepack: W (K=256 x N f32, row-major) -> Wt (N x 256 bf16) = W^T
// ---------------------------------------------------------------------------
__global__ void prepack_wt(const float* __restrict__ W, __bf16* __restrict__ Wt,
                           int N)
{
  int t = blockIdx.x * blockDim.x + threadIdx.x;
  if (t >= N * 256) return;
  const int n = t >> 8;
  const int k = t & 255;
  Wt[t] = (__bf16)W[(size_t)k * N + n];
}

// ---------------------------------------------------------------------------
// In-place softmax over contiguous groups of K (K = L*P <= 12)
// ---------------------------------------------------------------------------
__global__ void softmax_rows(float* __restrict__ aw, int rows, int K)
{
  int r = blockIdx.x * blockDim.x + threadIdx.x;
  if (r >= rows) return;
  float* p = aw + (size_t)r * K;
  float mx = p[0];
  for (int i = 1; i < K; ++i) mx = fmaxf(mx, p[i]);
  float e[12], s = 0.f;
  for (int i = 0; i < K; ++i) { e[i] = __expf(p[i] - mx); s += e[i]; }
  const float inv = 1.f / s;
  for (int i = 0; i < K; ++i) p[i] = e[i] * inv;
}

// ---------------------------------------------------------------------------
// Deformable sampling core. One thread = (n, q, h, dgroup-of-4-dims).
// value: (NB*S, 256) bf16 ; off: (NB*LQ, H*L*P*2) ; aw: (NB*LQ, H*L*P)
// Reference points: 32x32 grid centers (shared by all levels).
// grid_sample bilinear, align_corners=False, zero padding.
// ---------------------------------------------------------------------------
__global__ void deform_sample(const __bf16* __restrict__ value,
                              const float* __restrict__ off,
                              const float* __restrict__ aw,
                              float* __restrict__ out,
                              int nLevels, int S,
                              int e0, int e1, int e2,
                              int st0, int st1, int st2)
{
  int t = blockIdx.x * blockDim.x + threadIdx.x;
  if (t >= NB * LQ * N_HEADS * 4) return;
  const int dg = t & 3;
  const int h  = (t >> 2) & 15;
  const int q  = (t >> 6) & 1023;
  const int n  = t >> 16;

  const float gx = ((q & 31) + 0.5f) * (1.f / 32.f);
  const float gy = ((q >> 5) + 0.5f) * (1.f / 32.f);

  const int offStride = N_HEADS * nLevels * NPTS * 2;
  const int awStride  = N_HEADS * nLevels * NPTS;
  const float* offp = off + (size_t)(n * LQ + q) * offStride;
  const float* awp  = aw  + (size_t)(n * LQ + q) * awStride;

  const int es[3]  = { e0, e1, e2 };
  const int sts[3] = { st0, st1, st2 };

  float4 acc = { 0.f, 0.f, 0.f, 0.f };
  for (int l = 0; l < nLevels; ++l) {
    const int   e  = es[l];
    const float ef = (float)e;
    const int   base = sts[l];
    #pragma unroll
    for (int p = 0; p < NPTS; ++p) {
      const int idx = (h * nLevels + l) * NPTS + p;
      const float lx = gx + offp[idx * 2 + 0] / ef;
      const float ly = gy + offp[idx * 2 + 1] / ef;
      const float px = lx * ef - 0.5f;
      const float py = ly * ef - 0.5f;
      const float x0f = floorf(px), y0f = floorf(py);
      const float fx = px - x0f,  fy = py - y0f;
      const int x0 = (int)x0f, y0 = (int)y0f;
      const float a   = awp[idx];
      const float w00 = (1.f - fx) * (1.f - fy) * a;
      const float w10 = fx * (1.f - fy) * a;
      const float w01 = (1.f - fx) * fy * a;
      const float w11 = fx * fy * a;
      #define TAP(X, Y, W)                                                    \
        do {                                                                  \
          const int xi = (X), yi = (Y);                                       \
          if (xi >= 0 && xi < e && yi >= 0 && yi < e) {                       \
            const __bf16* vp = value +                                        \
              ((size_t)(n * S + base + yi * e + xi)) * 256 + h * 16 + dg * 4; \
            union { unsigned long long u; __bf16 hh[4]; } vv;                 \
            vv.u = *(const unsigned long long*)vp;                            \
            acc.x += (W) * (float)vv.hh[0];                                   \
            acc.y += (W) * (float)vv.hh[1];                                   \
            acc.z += (W) * (float)vv.hh[2];                                   \
            acc.w += (W) * (float)vv.hh[3];                                   \
          }                                                                   \
        } while (0)
      TAP(x0,     y0,     w00);
      TAP(x0 + 1, y0,     w10);
      TAP(x0,     y0 + 1, w01);
      TAP(x0 + 1, y0 + 1, w11);
      #undef TAP
    }
  }
  float* op = out + (size_t)(n * LQ + q) * 256 + h * 16 + dg * 4;
  op[0] = acc.x; op[1] = acc.y; op[2] = acc.z; op[3] = acc.w;
}

// ---------------------------------------------------------------------------
// out = query + gamma1 * attn + gamma2 * attn2   (per-column gammas)
// ---------------------------------------------------------------------------
__global__ void combine_out(const float* __restrict__ query,
                            const float* __restrict__ attn,
                            const float* __restrict__ attn2,
                            const float* __restrict__ g1,
                            const float* __restrict__ g2,
                            float* __restrict__ out, int n)
{
  int t = blockIdx.x * blockDim.x + threadIdx.x;
  if (t >= n) return;
  const int c = t & 255;
  out[t] = query[t] + g1[c] * attn[t] + g2[c] * attn2[t];
}

// ---------------------------------------------------------------------------
static inline size_t alignUp256(size_t x) { return (x + 255) & ~(size_t)255; }

extern "C" void kernel_launch(void* const* d_in, const int* in_sizes, int n_in,
                              void* d_out, int out_size, void* d_ws, size_t ws_size,
                              hipStream_t stream)
{
  (void)in_sizes; (void)n_in; (void)out_size; (void)ws_size;

  const float* srcsA[3] = { (const float*)d_in[0], (const float*)d_in[2], (const float*)d_in[4] };
  const float* srcsB[3] = { (const float*)d_in[1], (const float*)d_in[3], (const float*)d_in[5] };
  const float* qn_w   = (const float*)d_in[6];
  const float* qn_b   = (const float*)d_in[7];
  const float* qn_g   = (const float*)d_in[8];
  const float* qn_bt  = (const float*)d_in[9];
  const float* fn_w   = (const float*)d_in[10];
  const float* fn_b   = (const float*)d_in[11];
  const float* fn_g   = (const float*)d_in[12];
  const float* fn_bt  = (const float*)d_in[13];
  const float* n1_g   = (const float*)d_in[14];
  const float* n1_b   = (const float*)d_in[15];
  const float* gamma1 = (const float*)d_in[16];
  const float* gamma2 = (const float*)d_in[17];
  const float* ca_Wv  = (const float*)d_in[18];
  const float* ca_bv  = (const float*)d_in[19];
  const float* ca_Ws  = (const float*)d_in[20];
  const float* ca_bs  = (const float*)d_in[21];
  const float* ca_Wa  = (const float*)d_in[22];
  const float* ca_ba  = (const float*)d_in[23];
  const float* ca_Wo  = (const float*)d_in[24];
  const float* ca_bo  = (const float*)d_in[25];
  const float* sa_Wv  = (const float*)d_in[26];
  const float* sa_bv  = (const float*)d_in[27];
  const float* sa_Ws  = (const float*)d_in[28];
  const float* sa_bs  = (const float*)d_in[29];
  const float* sa_Wa  = (const float*)d_in[30];
  const float* sa_ba  = (const float*)d_in[31];
  const float* sa_Wo  = (const float*)d_in[32];
  const float* sa_bo  = (const float*)d_in[33];

  // workspace layout (~150 MB, reused across branches)
  char* p = (char*)d_ws;
  auto take = [&](size_t bytes) { char* r = p; p += alignUp256(bytes); return r; };
  __bf16* value    = (__bf16*)take((size_t)NB * S_TOT * 256 * 2);
  float*  off_ca   = (float*) take((size_t)NB * LQ * 384 * 4);
  float*  aw_ca    = (float*) take((size_t)NB * LQ * 192 * 4);
  float*  sampled  = (float*) take((size_t)NB * LQ * 256 * 4);
  float*  attn     = (float*) take((size_t)NB * LQ * 256 * 4);
  __bf16* value_sa = (__bf16*)take((size_t)NB * LQ * 256 * 2);
  float*  off_sa   = (float*) take((size_t)NB * LQ * 128 * 4);
  float*  aw_sa    = (float*) take((size_t)NB * LQ * 64 * 4);
  float*  sampled2 = (float*) take((size_t)NB * LQ * 256 * 4);
  float*  attn2    = (float*) take((size_t)NB * LQ * 256 * 4);
  // prepacked transposed bf16 weights (per-branch, ~1.1 MB, L2-resident)
  __bf16* pk_qn   = (__bf16*)take(256 * 256 * 2);
  __bf16* pk_fn   = (__bf16*)take(256 * 256 * 2);
  __bf16* pk_caWv = (__bf16*)take(256 * 256 * 2);
  __bf16* pk_caWs = (__bf16*)take(384 * 256 * 2);
  __bf16* pk_caWa = (__bf16*)take(192 * 256 * 2);
  __bf16* pk_caWo = (__bf16*)take(256 * 256 * 2);
  __bf16* pk_saWv = (__bf16*)take(256 * 256 * 2);
  __bf16* pk_saWs = (__bf16*)take(128 * 256 * 2);
  __bf16* pk_saWa = (__bf16*)take(64 * 256 * 2);
  __bf16* pk_saWo = (__bf16*)take(256 * 256 * 2);

  const int eArr[3]  = { 128, 64, 32 };
  const int stArr[3] = { 0, 16384, 20480 };

  for (int b = 0; b < 2; ++b) {
    const float* const* S = b ? srcsB : srcsA;
    #define SL(ptr, per) ((ptr) + (size_t)b * (per))
    Stage2 z = {};

    // 0) prepack this branch's weights to transposed bf16
    {
      auto pk = [&](const float* W, __bf16* Wt, int N) {
        prepack_wt<<<(N * 256 + 255) / 256, 256, 0, stream>>>(W, Wt, N);
      };
      pk(SL(qn_w, 256 * 256), pk_qn,   256);
      pk(SL(fn_w, 256 * 256), pk_fn,   256);
      pk(SL(ca_Wv, 256 * 256), pk_caWv, 256);
      pk(SL(ca_Ws, 256 * 384), pk_caWs, 384);
      pk(SL(ca_Wa, 256 * 192), pk_caWa, 192);
      pk(SL(ca_Wo, 256 * 256), pk_caWo, 256);
      pk(SL(sa_Wv, 256 * 256), pk_saWv, 256);
      pk(SL(sa_Ws, 256 * 128), pk_saWs, 128);
      pk(SL(sa_Wa, 256 * 64),  pk_saWa, 64);
      pk(SL(sa_Wo, 256 * 256), pk_saWo, 256);
    }

    // 1) q = LN(query @ qn_w + qn_b) -> off/aw projections (fused)
    {
      Stage2 sWs = { pk_caWs, SL(ca_bs, 384), off_ca, 384, 0 };
      Stage2 sWa = { pk_caWa, SL(ca_ba, 192), aw_ca,  192, 0 };
      fused_ln_gemm<<<NB * LQ / 16, 128, 0, stream>>>(
          S[2], pk_qn, SL(qn_b, 256), SL(qn_g, 256), SL(qn_bt, 256),
          nullptr, 1, sWs, sWa, z, 2, LQ, LQ, 0);
    }

    // 2) value = LN(feat @ fn_w + fn_b) @ ca_Wv + ca_bv   (fused, per level)
    for (int l = 0; l < 3; ++l) {
      Stage2 sWv = { pk_caWv, SL(ca_bv, 256), value, 256, 1 };
      const int rows = NB * eArr[l] * eArr[l];
      fused_ln_gemm<<<rows / 16, 128, 0, stream>>>(
          S[l], pk_fn, SL(fn_b, 256), SL(fn_g, 256), SL(fn_bt, 256),
          nullptr, 1, sWv, z, z, 1, eArr[l] * eArr[l], S_TOT, stArr[l]);
    }

    // 3) softmax over (L*P=12) attention weights
    softmax_rows<<<(NB * LQ * N_HEADS + 255) / 256, 256, 0, stream>>>(
        aw_ca, NB * LQ * N_HEADS, 12);

    // 4) multi-scale deformable sampling (per-batch value slice L2-resident)
    deform_sample<<<(NB * LQ * N_HEADS * 4 + 255) / 256, 256, 0, stream>>>(
        value, off_ca, aw_ca, sampled, 3, S_TOT, 128, 64, 32, 0, 16384, 20480);

    // 5) attn = sampled @ ca_Wo + ca_bo (written out);
    //    attn1 = LN(attn) -> sa value/off/aw projections (fused)
    {
      Stage2 sV  = { pk_saWv, SL(sa_bv, 256), value_sa, 256, 1 };
      Stage2 sS2 = { pk_saWs, SL(sa_bs, 128), off_sa,   128, 0 };
      Stage2 sA2 = { pk_saWa, SL(sa_ba, 64),  aw_sa,     64, 0 };
      fused_ln_gemm<<<NB * LQ / 16, 128, 0, stream>>>(
          sampled, pk_caWo, SL(ca_bo, 256), SL(n1_g, 256), SL(n1_b, 256),
          attn, 1, sV, sS2, sA2, 3, LQ, LQ, 0);
    }

    // 6) softmax over (1*P=4)
    softmax_rows<<<(NB * LQ * N_HEADS + 255) / 256, 256, 0, stream>>>(
        aw_sa, NB * LQ * N_HEADS, 4);

    // 7) single-level deformable self-attention sampling
    deform_sample<<<(NB * LQ * N_HEADS * 4 + 255) / 256, 256, 0, stream>>>(
        value_sa, off_sa, aw_sa, sampled2, 1, LQ, 32, 32, 32, 0, 0, 0);

    // 8) attn2 = sampled2 @ sa_Wo + sa_bo  (stage-1 only, no LN)
    fused_ln_gemm<<<NB * LQ / 16, 128, 0, stream>>>(
        sampled2, pk_saWo, SL(sa_bo, 256), nullptr, nullptr,
        attn2, 0, z, z, z, 0, LQ, LQ, 0);

    // 9) out = query + gamma1*attn + gamma2*attn2
    combine_out<<<(NB * LQ * 256 + 255) / 256, 256, 0, stream>>>(
        S[2], attn, attn2, SL(gamma1, 256), SL(gamma2, 256),
        (float*)d_out + (size_t)b * NB * LQ * 256, NB * LQ * 256);
    #undef SL
  }
}
